// Sggnn_for_test_50259707298052
// MI455X (gfx1250) — matrix operand order repaired and checked by reading, not compiled, not run on hardware
//
#include <hip/hip_runtime.h>
#include <hip/hip_bf16.h>

// ---------------------------------------------------------------------------
// Problem constants (from the reference)
// ---------------------------------------------------------------------------
constexpr int B_ = 128;
constexpr int G_ = 256;
constexpr int D_ = 512;
constexpr float BN_EPS_ = 1e-5f;
constexpr float RATIO_ = 0.1f;

typedef __attribute__((ext_vector_type(16))) __bf16 v16bf;
typedef __attribute__((ext_vector_type(8)))  float  v8f;

union FragU { v16bf v; unsigned int u[8]; };

// LDS row pitch in dwords for staged 64-row x 32-bf16 chunks.
// 20 dwords = 80 bytes: multiple of 16 (B128 async chunks stay aligned),
// stride-20 banking avoids half-wave fragment-read conflicts.
constexpr int PITCH_DW = 20;

// ---------------------------------------------------------------------------
// bf16 helpers (storage as ushort)
// ---------------------------------------------------------------------------
static __device__ __forceinline__ unsigned short f32_to_bf16(float f) {
    unsigned int u = __float_as_uint(f);
    unsigned int rounding = 0x7FFFu + ((u >> 16) & 1u);
    return (unsigned short)((u + rounding) >> 16);
}
static __device__ __forceinline__ float bf16_to_f32(unsigned short h) {
    return __uint_as_float(((unsigned int)h) << 16);
}

// CDNA5 async global->LDS copy, 16 bytes per lane (tracked by ASYNCcnt)
static __device__ __forceinline__ void async_copy_b128(unsigned lds_addr, const void* gaddr) {
    asm volatile("global_load_async_to_lds_b128 %0, %1, off"
                 :: "v"(lds_addr), "v"(gaddr) : "memory");
}
static __device__ __forceinline__ void wait_async0() {
    asm volatile("s_wait_asynccnt 0" ::: "memory");
}
// low 32 bits of a flat pointer into LDS == LDS byte address (aperture scheme)
static __device__ __forceinline__ unsigned lds_addr_of(const void* p) {
    return (unsigned)(unsigned long long)p;
}

// block-wide sum, blockDim.x == 256, sh has 256 floats
static __device__ __forceinline__ float block_reduce_sum(float v, float* sh) {
    int tid = threadIdx.x;
    sh[tid] = v;
    __syncthreads();
    for (int s = 128; s > 0; s >>= 1) {
        if (tid < s) sh[tid] += sh[tid + s];
        __syncthreads();
    }
    float r = sh[0];
    __syncthreads();
    return r;
}

// ---------------------------------------------------------------------------
// Kernel 0: convert W1/W2 -> bf16, wcd[j] = Wc[1,j]-Wc[0,j]
// ---------------------------------------------------------------------------
__global__ void convert_kernel(const float* __restrict__ W1,
                               const float* __restrict__ W2,
                               const float* __restrict__ Wc,
                               unsigned short* __restrict__ w1b,
                               unsigned short* __restrict__ w2b,
                               float* __restrict__ wcd) {
    int i = blockIdx.x * 256 + threadIdx.x;
    if (i < D_ * D_) {
        w1b[i] = f32_to_bf16(W1[i]);
        w2b[i] = f32_to_bf16(W2[i]);
    }
    if (i < D_) {
        wcd[i] = Wc[D_ + i] - Wc[i];
    }
}

// ---------------------------------------------------------------------------
// Kernel 1: per (b,g) row: d = (qf-gf)^2 -> dbf (bf16), gf -> gfb (bf16),
//           sq[bg] = sum gf^2, dcd[bg] = sum d * wcd
// grid(G_, B_), block 256
// ---------------------------------------------------------------------------
__global__ void prep_kernel(const float* __restrict__ qf,
                            const float* __restrict__ gf,
                            const float* __restrict__ wcd,
                            unsigned short* __restrict__ dbf,
                            unsigned short* __restrict__ gfb,
                            float* __restrict__ sq,
                            float* __restrict__ dcd) {
    __shared__ float r1[256];
    __shared__ float r2[256];
    const int g = blockIdx.x, b = blockIdx.y;
    const size_t row = (size_t)b * G_ + g;
    const size_t base = row * D_;
    float sq_acc = 0.f, dc_acc = 0.f;
#pragma unroll
    for (int i = 0; i < 2; ++i) {
        int e = threadIdx.x + i * 256;
        float qv = qf[(size_t)b * D_ + e];
        float gv = gf[base + e];
        float df = qv - gv;
        float d2 = df * df;
        dbf[base + e] = f32_to_bf16(d2);
        gfb[base + e] = f32_to_bf16(gv);
        sq_acc += gv * gv;
        dc_acc += d2 * wcd[e];
    }
    float s1 = block_reduce_sum(sq_acc, r1);
    float s2 = block_reduce_sum(dc_acc, r2);
    if (threadIdx.x == 0) { sq[row] = s1; dcd[row] = s2; }
}

// ---------------------------------------------------------------------------
// Shared WMMA GEMM tile: C[64x64] (f32) = A[rowBase..+64,:K] . B[colBase..+64,:K]^T
// A, B bf16 row-major, row length K. 256 threads = 8 waves in a 4x2 grid; each
// wave owns a 16x32 C strip (two 16x16 WMMA tiles, A fragment reused).
// Global->LDS staging uses async B128 copies with double buffering.
// ---------------------------------------------------------------------------
static __device__ __forceinline__ void gemm_nt_tile64(const unsigned short* __restrict__ Aq,
                                                      const unsigned short* __restrict__ Bq,
                                                      int K, int rowBase, int colBase,
                                                      v8f& acc0, v8f& acc1) {
    __shared__ __align__(16) unsigned int lds[2][2][64 * PITCH_DW];

    const int tid   = threadIdx.x;
    const int wave  = tid >> 5;
    const int lane  = tid & 31;
    const int waveM = wave >> 1;   // 0..3  (16-row subtile)
    const int waveN = wave & 1;    // 0..1  (32-col strip)
    const int half  = lane >> 4;   // 0/1
    const int lr    = lane & 15;

    // staging assignment: thread -> (row, 16B chunk) of the 64x(32 bf16) chunk
    const int crow = tid >> 2;     // 0..63
    const int c4   = tid & 3;      // 0..3 (which 16B piece of the 64B row)
    const char* gA = (const char*)Aq + (size_t)(rowBase + crow) * K * 2 + c4 * 16;
    const char* gB = (const char*)Bq + (size_t)(colBase + crow) * K * 2 + c4 * 16;
    const unsigned ldsOff = (unsigned)(crow * (PITCH_DW * 4) + c4 * 16);
    unsigned baseA[2], baseB[2];
    baseA[0] = lds_addr_of(&lds[0][0][0]);
    baseA[1] = lds_addr_of(&lds[1][0][0]);
    baseB[0] = lds_addr_of(&lds[0][1][0]);
    baseB[1] = lds_addr_of(&lds[1][1][0]);

    // prologue: stage K-chunk 0 into buffer 0
    async_copy_b128(baseA[0] + ldsOff, gA);
    async_copy_b128(baseB[0] + ldsOff, gB);
    wait_async0();
    __syncthreads();

    acc0 = (v8f){0.f, 0.f, 0.f, 0.f, 0.f, 0.f, 0.f, 0.f};
    acc1 = (v8f){0.f, 0.f, 0.f, 0.f, 0.f, 0.f, 0.f, 0.f};

    const int nsteps = K >> 5;     // K-chunks of 32
    for (int s = 0; s < nsteps; ++s) {
        const int cur = s & 1;
        if (s + 1 < nsteps) {
            // issue next chunk's async copies into the other buffer (overlap)
            const char* nA = gA + (size_t)(s + 1) * 64;   // 32 bf16 = 64 bytes
            const char* nB = gB + (size_t)(s + 1) * 64;
            async_copy_b128(baseA[cur ^ 1] + ldsOff, nA);
            async_copy_b128(baseB[cur ^ 1] + ldsOff, nB);
        }
        const unsigned int* As = &lds[cur][0][0];
        const unsigned int* Bs = &lds[cur][1][0];
        FragU a, b0, b1;
#pragma unroll
        for (int v = 0; v < 8; ++v) {
            // A 16x32 bf16 fragment: lane lr = row M; v0..3 K base half*8,
            // v4..7 K base 16+half*8 (packed bf16 pairs -> dword index)
            int ka = (v < 4 ? 0 : 8) + half * 4 + (v & 3);
            a.u[v]  = As[(waveM * 16 + lr) * PITCH_DW + ka];
            // B 32x16 bf16 fragment: column N = lr; same K layout as A
            // (v0..3 K base half*8, v4..7 K base 16+half*8)
            int kb = (v < 4 ? 0 : 8) + half * 4 + (v & 3);
            b0.u[v] = Bs[(waveN * 32 + lr) * PITCH_DW + kb];
            b1.u[v] = Bs[(waveN * 32 + 16 + lr) * PITCH_DW + kb];
        }
        acc0 = __builtin_amdgcn_wmma_f32_16x16x32_bf16(false, a.v, false, b0.v,
                                                       (short)0, acc0, false, false);
        acc1 = __builtin_amdgcn_wmma_f32_16x16x32_bf16(false, a.v, false, b1.v,
                                                       (short)0, acc1, false, false);
        wait_async0();
        __syncthreads();
    }
}

// ---------------------------------------------------------------------------
// Kernel 2: Gram -> masked-w:  wm[b,i,j] = (i==j) ? -99 : exp(-(sq_i+sq_j-2*gram))
// grid(G/64, G/64, B_), block 256
// ---------------------------------------------------------------------------
__global__ void gram_kernel(const unsigned short* __restrict__ gfb,
                            const float* __restrict__ sq,
                            float* __restrict__ wm) {
    const int b = blockIdx.z;
    const int rowBase = blockIdx.x * 64;
    const int colBase = blockIdx.y * 64;
    const unsigned short* Gp = gfb + (size_t)b * G_ * D_;

    v8f acc0, acc1;
    gemm_nt_tile64(Gp, Gp, D_, rowBase, colBase, acc0, acc1);

    const int lane = threadIdx.x & 31;
    const int wave = threadIdx.x >> 5;
    const int half = lane >> 4, nc = lane & 15;
    const int rowT = rowBase + (wave >> 1) * 16;
    const int colT = colBase + (wave & 1) * 32;
    const float* sqb = sq + (size_t)b * G_;
#pragma unroll
    for (int v = 0; v < 8; ++v) {
        int i = rowT + half * 8 + v;
        int j0 = colT + nc;
        int j1 = colT + 16 + nc;
        float d0 = sqb[i] + sqb[j0] - 2.0f * acc0[v];
        float d1 = sqb[i] + sqb[j1] - 2.0f * acc1[v];
        wm[((size_t)b * G_ + i) * G_ + j0] = (i == j0) ? -99.0f : expf(-d0);
        wm[((size_t)b * G_ + i) * G_ + j1] = (i == j1) ? -99.0f : expf(-d1);
    }
}

// ---------------------------------------------------------------------------
// Kernel 3: fused Linear + BN(eval) + LeakyReLU(0.1) -> bf16
//   out[m,n] = lrelu( (A[m,:].W[n,:] + bvec[n] - rm[n]) * g[n]/sqrt(rv[n]+eps) + be[n] )
// grid(M/64, D/64), block 256.  M = B_*G_, N = K = D_.
// ---------------------------------------------------------------------------
__global__ void refine_gemm_kernel(const unsigned short* __restrict__ A,
                                   const unsigned short* __restrict__ W,
                                   const float* __restrict__ bvec,
                                   const float* __restrict__ gvec,
                                   const float* __restrict__ bevec,
                                   const float* __restrict__ rmvec,
                                   const float* __restrict__ rvvec,
                                   unsigned short* __restrict__ out) {
    const int rowBase = blockIdx.x * 64;
    const int colBase = blockIdx.y * 64;

    v8f acc0, acc1;
    gemm_nt_tile64(A, W, D_, rowBase, colBase, acc0, acc1);

    const int lane = threadIdx.x & 31;
    const int wave = threadIdx.x >> 5;
    const int half = lane >> 4, nc = lane & 15;
    const int rowT = rowBase + (wave >> 1) * 16;
    const int colT = colBase + (wave & 1) * 32;

    const int n0 = colT + nc;
    const int n1 = colT + 16 + nc;
    const float sc0 = gvec[n0] / sqrtf(rvvec[n0] + BN_EPS_);
    const float sh0 = (bvec[n0] - rmvec[n0]) * sc0 + bevec[n0];
    const float sc1 = gvec[n1] / sqrtf(rvvec[n1] + BN_EPS_);
    const float sh1 = (bvec[n1] - rmvec[n1]) * sc1 + bevec[n1];
#pragma unroll
    for (int v = 0; v < 8; ++v) {
        int m = rowT + half * 8 + v;
        float h0 = acc0[v] * sc0 + sh0;
        float h1 = acc1[v] * sc1 + sh1;
        h0 = (h0 > 0.f) ? h0 : 0.1f * h0;
        h1 = (h1 > 0.f) ? h1 : 0.1f * h1;
        out[(size_t)m * D_ + n0] = f32_to_bf16(h0);
        out[(size_t)m * D_ + n1] = f32_to_bf16(h1);
    }
}

// ---------------------------------------------------------------------------
// Kernel 4: tcd[row] = sum_j t[row,j] * wcd[j]   (row = b*G+m)
// grid(B_*G_), block 256
// ---------------------------------------------------------------------------
__global__ void tcd_kernel(const unsigned short* __restrict__ tbf,
                           const float* __restrict__ wcd,
                           float* __restrict__ tcd) {
    __shared__ float r1[256];
    const size_t row = blockIdx.x;
    float a = 0.f;
#pragma unroll
    for (int i = 0; i < 2; ++i) {
        int e = threadIdx.x + i * 256;
        a += bf16_to_f32(tbf[row * D_ + e]) * wcd[e];
    }
    float s = block_reduce_sum(a, r1);
    if (threadIdx.x == 0) tcd[row] = s;
}

// ---------------------------------------------------------------------------
// Kernel 5: column softmax of wm over m, fused with score
//   s[b,k] = RATIO * (sum_m tcd[b,m]*e_m)/(sum_m e_m) + (1-RATIO)*dcd[b,k] + (bc1-bc0)
// grid(G_, B_), block 256 (thread = m)
// ---------------------------------------------------------------------------
__global__ void softmax_score_kernel(const float* __restrict__ wm,
                                     const float* __restrict__ tcd,
                                     const float* __restrict__ dcd,
                                     const float* __restrict__ bc,
                                     float* __restrict__ sres) {
    __shared__ float r1[256];
    __shared__ float r2[256];
    const int k = blockIdx.x, b = blockIdx.y;
    const int m = threadIdx.x;
    // entries bounded above by ~1 (exp(-dist2) <= ~1, diag = -99): no max pass needed
    float e  = expf(wm[((size_t)b * G_ + m) * G_ + k]);
    float te = e * tcd[(size_t)b * G_ + m];
    float se  = block_reduce_sum(e, r1);
    float ste = block_reduce_sum(te, r2);
    if (threadIdx.x == 0) {
        float s = RATIO_ * (ste / se) + (1.0f - RATIO_) * dcd[(size_t)b * G_ + k]
                + (bc[1] - bc[0]);
        sres[(size_t)b * G_ + k] = s;
    }
}

// ---------------------------------------------------------------------------
// Kernel 6: per-batch stable argsort (descending) of 256 scores -> int indices
// grid(B_), block 256; bitonic sort with index tie-break (emulates stable sort)
// ---------------------------------------------------------------------------
__global__ void argsort_kernel(const float* __restrict__ sres,
                               int* __restrict__ outIdx) {
    __shared__ float sv[256];
    __shared__ int   si[256];
    const int b = blockIdx.x;
    const int tid = threadIdx.x;
    sv[tid] = sres[(size_t)b * G_ + tid];
    si[tid] = tid;
    __syncthreads();
    for (int k = 2; k <= 256; k <<= 1) {
        for (int j = k >> 1; j > 0; j >>= 1) {
            int ixj = tid ^ j;
            if (ixj > tid) {
                bool ascending = ((tid & k) != 0);  // final pass (k=256): descending
                float v0 = sv[tid], v1 = sv[ixj];
                int   i0 = si[tid], i1 = si[ixj];
                bool swap_;
                if (ascending) swap_ = (v0 > v1) || (v0 == v1 && i0 < i1);
                else           swap_ = (v0 < v1) || (v0 == v1 && i0 > i1);
                if (swap_) {
                    sv[tid] = v1; sv[ixj] = v0;
                    si[tid] = i1; si[ixj] = i0;
                }
            }
            __syncthreads();
        }
    }
    outIdx[(size_t)b * G_ + tid] = si[tid];
}

// ---------------------------------------------------------------------------
// Host-side orchestration
// ---------------------------------------------------------------------------
static inline size_t align256(size_t x) { return (x + 255) & ~(size_t)255; }

extern "C" void kernel_launch(void* const* d_in, const int* in_sizes, int n_in,
                              void* d_out, int out_size, void* d_ws, size_t ws_size,
                              hipStream_t stream) {
    if (n_in < 16) return;
    const float* qf  = (const float*)d_in[0];
    const float* gf  = (const float*)d_in[1];
    const float* W1  = (const float*)d_in[2];
    const float* b1  = (const float*)d_in[3];
    const float* g1  = (const float*)d_in[4];
    const float* be1 = (const float*)d_in[5];
    const float* rm1 = (const float*)d_in[6];
    const float* rv1 = (const float*)d_in[7];
    const float* W2  = (const float*)d_in[8];
    const float* b2  = (const float*)d_in[9];
    const float* g2  = (const float*)d_in[10];
    const float* be2 = (const float*)d_in[11];
    const float* rm2 = (const float*)d_in[12];
    const float* rv2 = (const float*)d_in[13];
    const float* Wc  = (const float*)d_in[14];
    const float* bc  = (const float*)d_in[15];
    int* outIdx = (int*)d_out;  // argsort indices (int32)

    // Workspace carving
    char* ws = (char*)d_ws;
    size_t off = 0;
    const size_t BG  = (size_t)B_ * G_;
    const size_t BGD = BG * D_;

    unsigned short* gfb = (unsigned short*)(ws + off); off = align256(off + BGD * 2);
    unsigned short* dbf = (unsigned short*)(ws + off); off = align256(off + BGD * 2); // reused as t
    unsigned short* h1b = (unsigned short*)(ws + off); off = align256(off + BGD * 2);
    unsigned short* w1b = (unsigned short*)(ws + off); off = align256(off + (size_t)D_ * D_ * 2);
    unsigned short* w2b = (unsigned short*)(ws + off); off = align256(off + (size_t)D_ * D_ * 2);
    float* wcd  = (float*)(ws + off); off = align256(off + (size_t)D_ * 4);
    float* sq   = (float*)(ws + off); off = align256(off + BG * 4);
    float* dcd  = (float*)(ws + off); off = align256(off + BG * 4);
    float* tcd  = (float*)(ws + off); off = align256(off + BG * 4);
    float* sres = (float*)(ws + off); off = align256(off + BG * 4);
    float* wm   = (float*)(ws + off); off = align256(off + BG * G_ * 4);
    (void)ws_size; (void)in_sizes; (void)out_size;

    // 0) weight conversion + classifier diff vector
    convert_kernel<<<dim3((D_ * D_ + 255) / 256), dim3(256), 0, stream>>>(
        W1, W2, Wc, w1b, w2b, wcd);

    // 1) d, gf -> bf16; sq; dcd
    prep_kernel<<<dim3(G_, B_), dim3(256), 0, stream>>>(
        qf, gf, wcd, dbf, gfb, sq, dcd);

    // 2) batched Gram -> masked-w (WMMA bf16, async-LDS staged)
    gram_kernel<<<dim3(G_ / 64, G_ / 64, B_), dim3(256), 0, stream>>>(gfb, sq, wm);

    // 3) refine layer 1: h = lrelu(BN(d @ W1^T + b1))
    refine_gemm_kernel<<<dim3((unsigned)(BG / 64), D_ / 64), dim3(256), 0, stream>>>(
        dbf, w1b, b1, g1, be1, rm1, rv1, h1b);

    // 4) refine layer 2: t = lrelu(BN(h @ W2^T + b2)), t aliases dbf buffer
    refine_gemm_kernel<<<dim3((unsigned)(BG / 64), D_ / 64), dim3(256), 0, stream>>>(
        h1b, w2b, b2, g2, be2, rm2, rv2, dbf);

    // 5) tcd[row] = t[row,:] . wcd
    tcd_kernel<<<dim3((unsigned)BG), dim3(256), 0, stream>>>(dbf, wcd, tcd);

    // 6) column softmax + score
    softmax_score_kernel<<<dim3(G_, B_), dim3(256), 0, stream>>>(
        wm, tcd, dcd, bc, sres);

    // 7) per-batch stable descending argsort
    argsort_kernel<<<dim3(B_), dim3(256), 0, stream>>>(sres, outIdx);
}
